// GraphNet_5841155522636
// MI455X (gfx1250) — matrix operand order repaired and checked by reading, not compiled, not run on hardware
//
#include <hip/hip_runtime.h>
#include <hip/hip_bf16.h>

// Problem constants (match reference)
#define NN   100000
#define NE   1600000
#define INF_ 32
#define DD   128
#define OUTF 16
#define MT   (NN / 16)          // 6250 M-tiles, exact; MT/2 = 3125 exact

typedef __attribute__((ext_vector_type(16))) __bf16 v16bf;
typedef __attribute__((ext_vector_type(8)))  __bf16 v8bf;
typedef __attribute__((ext_vector_type(8)))  float  v8f;

// ---------------------------------------------------------------------------
// WMMA fragment helpers (CDNA5 v_wmma_f32_16x16x32_bf16, wave32)
// ---------------------------------------------------------------------------

// A-fragment from row-major bf16: lane holds row = lane&15;
// halves 0..7 = K kb..kb+7, halves 8..15 = K kb+16..kb+23 (kb = k0 + (lane>>4)*8).
// p is the lane-resolved pointer; two contiguous 16 B loads.
__device__ __forceinline__ v16bf load_a_bf16(const __bf16* __restrict__ p) {
    v8bf lo = *(const v8bf*)(p);
    v8bf hi = *(const v8bf*)(p + 16);
    return __builtin_shufflevector(lo, hi, 0,1,2,3,4,5,6,7,8,9,10,11,12,13,14,15);
}

// B-fragment from packed fragment-major weights: frag = kt*ntiles + nt,
// lane's 16 halves stored contiguously -> one 32 B per-lane load.
__device__ __forceinline__ v16bf load_b_packed(const __bf16* __restrict__ wp,
                                               int frag, int lane) {
    return *(const v16bf*)(wp + ((size_t)frag << 9) + lane * 16);
}

// D-fragment store: VGPR r -> row = row0 + r + (lane>>4)*8, col = col0 + (lane&15).
__device__ __forceinline__ void store_d(float* __restrict__ base, int ld,
                                        int row0, int col0, int lane, v8f c, int relu) {
    const int col   = col0 + (lane & 15);
    const int rbase = row0 + ((lane >> 4) << 3);
#pragma unroll
    for (int r = 0; r < 8; ++r) {
        float v = c[r];
        if (relu) v = fmaxf(v, 0.0f);
        base[(size_t)(rbase + r) * ld + col] = v;
    }
}

__device__ __forceinline__ v8f bias_acc(const float* __restrict__ bias, int col0, int lane) {
    const float bv = bias[col0 + (lane & 15)];
    v8f c;
#pragma unroll
    for (int r = 0; r < 8; ++r) c[r] = bv;
    return c;
}

// ---------------------------------------------------------------------------
// Pre-pass kernels
// ---------------------------------------------------------------------------

// Pack a K x N fp32 weight (row-major) into fragment-major bf16:
// out[((kt*ntiles + nt)*32 + lane)*16 + e] = W[kt*32 + (lane>>4)*8 + (e&7) + (e>>3)*16][nt*16 + (lane&15)]
__global__ void pack_w(const float* __restrict__ W, int K, int N, __bf16* __restrict__ out) {
    int i = blockIdx.x * blockDim.x + threadIdx.x;
    if (i >= K * N) return;
    int frag = i >> 9;
    int rem  = i & 511;
    int lane = rem >> 4;
    int e    = rem & 15;
    int ntiles = N >> 4;
    int nt = frag % ntiles;
    int kt = frag / ntiles;
    int col = nt * 16 + (lane & 15);
    int k   = kt * 32 + ((lane >> 4) << 3) + (e & 7) + ((e >> 3) << 4);
    out[i] = (__bf16)W[(size_t)k * N + col];
}

// fp32 -> bf16 for node features x
__global__ void cvt_x_kernel(const float* __restrict__ x, __bf16* __restrict__ xbf) {
    int i = blockIdx.x * blockDim.x + threadIdx.x;
    if (i < NN * INF_) xbf[i] = (__bf16)x[i];
}

// Build K=256 concatenated activation rows: act[n][0..127] = agg[n]*invc[n],
// act[n][128..255] = h[n]  (single cvt pass per layer).
__global__ void cvt_layer(const float* __restrict__ agg, const float* __restrict__ invc,
                          const float* __restrict__ h, __bf16* __restrict__ act) {
    int i = blockIdx.x * blockDim.x + threadIdx.x;
    if (i >= NN * DD) return;
    int n = i >> 7, d = i & 127;
    act[(size_t)n * 256 + d]        = (__bf16)(agg[i] * invc[n]);
    act[(size_t)n * 256 + 128 + d]  = (__bf16)h[i];
}

// Final conv output -> bf16 (into the "h" half of act rows) for the head GEMM.
__global__ void cvt_hfinal(const float* __restrict__ h, __bf16* __restrict__ act) {
    int i = blockIdx.x * blockDim.x + threadIdx.x;
    if (i >= NN * DD) return;
    int n = i >> 7, d = i & 127;
    act[(size_t)n * 256 + 128 + d] = (__bf16)h[i];
}

// In-degree count: one thread per edge (graph is shared by both conv layers)
__global__ void degree_kernel(const int* __restrict__ dst, float* __restrict__ cnt) {
    int e = blockIdx.x * blockDim.x + threadIdx.x;
    if (e < NE) unsafeAtomicAdd(cnt + dst[e], 1.0f);
}

__global__ void invc_kernel(const float* __restrict__ cnt, float* __restrict__ invc) {
    int n = blockIdx.x * blockDim.x + threadIdx.x;
    if (n < NN) invc[n] = 1.0f / fmaxf(cnt[n], 1.0f);
}

// Edge scatter: one wave per edge; lane handles 4 consecutive features (float4).
// h is 51 MB -> L2-resident on MI455X (192 MB L2): gather + atomics stay in L2.
__global__ void scatter_kernel(const int* __restrict__ src, const int* __restrict__ dst,
                               const float* __restrict__ h, float* __restrict__ agg) {
    int wave = (blockIdx.x * blockDim.x + threadIdx.x) >> 5;
    int lane = threadIdx.x & 31;
    if (wave >= NE) return;
    const int s = src[wave];
    const int d = dst[wave];
    const float4 v = ((const float4*)(h + (size_t)s * DD))[lane];
    float* ap = agg + (size_t)d * DD + lane * 4;
    unsafeAtomicAdd(ap + 0, v.x);
    unsafeAtomicAdd(ap + 1, v.y);
    unsafeAtomicAdd(ap + 2, v.z);
    unsafeAtomicAdd(ap + 3, v.w);
}

// ---------------------------------------------------------------------------
// GEMM kernels
// ---------------------------------------------------------------------------

// Node embedding: h = x @ W_emb + b_emb  ([N,32]x[32,128]); 1 wmma/wave.
__global__ void embed_gemm(const __bf16* __restrict__ xbf, const __bf16* __restrict__ wp,
                           const float* __restrict__ bias, float* __restrict__ h) {
    const int nt   = threadIdx.x >> 5;
    const int lane = threadIdx.x & 31;
    const int row0 = blockIdx.x * 16;
    const int r0   = row0 + (lane & 15);
    v8f acc = bias_acc(bias, nt * 16, lane);
    v16bf a = load_a_bf16(xbf + (size_t)r0 * INF_ + ((lane >> 4) << 3));
    v16bf b = load_b_packed(wp, nt, lane);
    acc = __builtin_amdgcn_wmma_f32_16x16x32_bf16(false, a, false, b, (short)0, acc,
                                                  false, false);
    store_d(h, DD, row0, nt * 16, lane, acc, 0);
}

// Fused RGCN layer as one K=256 GEMM over concatenated activations:
// out = relu(act @ [W_rel; W_root] + b). Block = 8 waves (one N-tile each),
// each wave computes 2 M-tiles sharing B-fragments -> 16 wmma/wave.
__global__ void rgcn_gemm(const __bf16* __restrict__ act, const __bf16* __restrict__ wp,
                          const float* __restrict__ bias, float* __restrict__ out) {
    const int nt    = threadIdx.x >> 5;
    const int lane  = threadIdx.x & 31;
    const int row0  = blockIdx.x * 32;
    const int r0    = row0 + (lane & 15);
    const int kboff = (lane >> 4) << 3;
    v8f acc0 = bias_acc(bias, nt * 16, lane);
    v8f acc1 = acc0;
#pragma unroll
    for (int kt = 0; kt < 8; ++kt) {
        const int kb = kt * 32 + kboff;
        v16bf a0 = load_a_bf16(act + (size_t)r0 * 256 + kb);
        v16bf a1 = load_a_bf16(act + (size_t)(r0 + 16) * 256 + kb);
        v16bf b  = load_b_packed(wp, kt * 8 + nt, lane);
        acc0 = __builtin_amdgcn_wmma_f32_16x16x32_bf16(false, a0, false, b, (short)0, acc0,
                                                       false, false);
        acc1 = __builtin_amdgcn_wmma_f32_16x16x32_bf16(false, a1, false, b, (short)0, acc1,
                                                       false, false);
    }
    store_d(out, DD, row0,      nt * 16, lane, acc0, 1);
    store_d(out, DD, row0 + 16, nt * 16, lane, acc1, 1);
}

// Head: d_out = h @ W_out + b_out  ([N,128]x[128,16]); A read from the "h" half
// of the concatenated act rows (offset 128, ld 256). 4 wmma/wave.
__global__ void head_gemm(const __bf16* __restrict__ act, const __bf16* __restrict__ wp,
                          const float* __restrict__ bias, float* __restrict__ out) {
    const int mtile = blockIdx.x * 8 + (threadIdx.x >> 5);
    const int lane  = threadIdx.x & 31;
    if (mtile >= MT) return;                      // wave-uniform guard
    const int r0    = mtile * 16 + (lane & 15);
    const int kboff = (lane >> 4) << 3;
    v8f acc = bias_acc(bias, 0, lane);
#pragma unroll
    for (int kt = 0; kt < 4; ++kt) {
        v16bf a = load_a_bf16(act + (size_t)r0 * 256 + 128 + kt * 32 + kboff);
        v16bf b = load_b_packed(wp, kt, lane);
        acc = __builtin_amdgcn_wmma_f32_16x16x32_bf16(false, a, false, b, (short)0, acc,
                                                      false, false);
    }
    store_d(out, OUTF, mtile * 16, 0, lane, acc, 0);
}

// ---------------------------------------------------------------------------
// Host-side launch
// ---------------------------------------------------------------------------
extern "C" void kernel_launch(void* const* d_in, const int* in_sizes, int n_in,
                              void* d_out, int out_size, void* d_ws, size_t ws_size,
                              hipStream_t stream) {
    // Inputs per setup_inputs() order
    const float* x      = (const float*)d_in[0];
    const int*   eidx   = (const int*)  d_in[1];   // [2, E]: row 0 = src, row 1 = dst
    const float* W_emb  = (const float*)d_in[2];
    const float* b_emb  = (const float*)d_in[3];
    const float* W1_rel = (const float*)d_in[4];
    const float* W1_root= (const float*)d_in[5];
    const float* b1     = (const float*)d_in[6];
    const float* W2_rel = (const float*)d_in[7];
    const float* W2_root= (const float*)d_in[8];
    const float* b2     = (const float*)d_in[9];
    const float* W_out  = (const float*)d_in[10];
    const float* b_out  = (const float*)d_in[11];
    const int* src = eidx;
    const int* dst = eidx + NE;

    // Workspace layout (fp32 first, then bf16; all offsets keep >=32 B alignment)
    float* h    = (float*)d_ws;              // N*D fp32
    float* h2   = h    + (size_t)NN * DD;    // N*D fp32
    float* agg  = h2   + (size_t)NN * DD;    // N*D fp32
    float* cnt  = agg  + (size_t)NN * DD;    // N fp32
    float* invc = cnt  + NN;                 // N fp32
    __bf16* actbf  = (__bf16*)(invc + NN);   // N*256 bf16 (concat [agg_norm | h])
    __bf16* xbf    = actbf  + (size_t)NN * 256;  // N*32 bf16
    __bf16* Wemb_p = xbf    + (size_t)NN * INF_; // 32*128 packed
    __bf16* Wcat1  = Wemb_p + INF_ * DD;         // 256*128 packed ([W1_rel; W1_root])
    __bf16* Wcat2  = Wcat1  + 2 * DD * DD;       // 256*128 packed
    __bf16* Wout_p = Wcat2  + 2 * DD * DD;       // 128*16 packed

    // 1) pack weights into fragment-major bf16 (tiny one-off work)
    pack_w<<<(INF_*DD + 255)/256, 256, 0, stream>>>(W_emb,   INF_, DD,  Wemb_p);
    pack_w<<<(DD*DD  + 255)/256, 256, 0, stream>>>(W1_rel,  DD,   DD,  Wcat1);
    pack_w<<<(DD*DD  + 255)/256, 256, 0, stream>>>(W1_root, DD,   DD,  Wcat1 + DD*DD);
    pack_w<<<(DD*DD  + 255)/256, 256, 0, stream>>>(W2_rel,  DD,   DD,  Wcat2);
    pack_w<<<(DD*DD  + 255)/256, 256, 0, stream>>>(W2_root, DD,   DD,  Wcat2 + DD*DD);
    pack_w<<<(DD*OUTF+ 255)/256, 256, 0, stream>>>(W_out,   DD,   OUTF, Wout_p);

    // 2) degree counts -> invc (shared by both conv layers)
    hipMemsetAsync(cnt, 0, (size_t)NN * sizeof(float), stream);
    degree_kernel<<<(NE + 255)/256, 256, 0, stream>>>(dst, cnt);
    invc_kernel<<<(NN + 255)/256, 256, 0, stream>>>(cnt, invc);

    // 3) embedding: h = x @ W_emb + b_emb
    cvt_x_kernel<<<(NN*INF_ + 255)/256, 256, 0, stream>>>(x, xbf);
    embed_gemm<<<MT, 256, 0, stream>>>(xbf, Wemb_p, b_emb, h);

    // 4) conv1: scatter-sum h -> agg; build act = [agg/cnt | h]; GEMM -> h2 (relu)
    hipMemsetAsync(agg, 0, (size_t)NN * DD * sizeof(float), stream);
    scatter_kernel<<<(NE * 32 + 255)/256, 256, 0, stream>>>(src, dst, h, agg);
    cvt_layer<<<(NN*DD + 255)/256, 256, 0, stream>>>(agg, invc, h, actbf);
    rgcn_gemm<<<MT/2, 256, 0, stream>>>(actbf, Wcat1, b1, h2);

    // 5) conv2: scatter-sum h2 -> agg; act = [agg/cnt | h2]; GEMM -> h (relu)
    hipMemsetAsync(agg, 0, (size_t)NN * DD * sizeof(float), stream);
    scatter_kernel<<<(NE * 32 + 255)/256, 256, 0, stream>>>(src, dst, h2, agg);
    cvt_layer<<<(NN*DD + 255)/256, 256, 0, stream>>>(agg, invc, h2, actbf);
    rgcn_gemm<<<MT/2, 256, 0, stream>>>(actbf, Wcat2, b2, h);

    // 6) head: d_out = h @ W_out + b_out
    cvt_hfinal<<<(NN*DD + 255)/256, 256, 0, stream>>>(h, actbf);
    head_gemm<<<(MT + 7)/8, 256, 0, stream>>>(actbf, Wout_p, b_out, (float*)d_out);
}
